// SignCondPackedBitLinear_83880711291073
// MI455X (gfx1250) — matrix operand order repaired and checked by reading, not compiled, not run on hardware
//
#include <hip/hip_runtime.h>
#include <hip/hip_bf16.h>
#include <stddef.h>

// Problem constants (from reference): out = x @ W^T
//   x: [B*S, K] = [8192, 4096] f32
//   signs: [N, K] = [11008, 4096] f32 (+-1)
//   scale_pos/scale_neg: [N*K/128] f32, group g = n*32 + k/128
#define M_TOT 8192
#define N_TOT 11008
#define K_TOT 4096
#define GSZ   128

#define BM 128
#define BN 128
#define BK 64
#define LDP 72                 // padded LDS row (bf16 elements)
#define NSTEPS (K_TOT / BK)    // 64

typedef __attribute__((ext_vector_type(16))) __bf16 v16bf;
typedef __attribute__((ext_vector_type(8)))  float  v8f;
typedef __attribute__((ext_vector_type(8)))  unsigned short v8u16;
typedef __attribute__((ext_vector_type(16))) unsigned short v16u16;
typedef __attribute__((ext_vector_type(4)))  unsigned short v4u16;

// f32 -> bf16 via native FPTrunc (backend picks v_cvt hardware; RNE).
static __device__ __forceinline__ unsigned short cvt_bf16(float f) {
  __bf16 h = (__bf16)f;
  return __builtin_bit_cast(unsigned short, h);
}

// Load a 16-element bf16 fragment from LDS as two 16-byte chunks.
// 'step' is the element distance between the two chunks (A: 16, B: 16).
static __device__ __forceinline__ v16bf frag_ld(const unsigned short* p, int step) {
  v8u16 lo = *(const v8u16*)p;
  v8u16 hi = *(const v8u16*)(p + step);
  v16u16 w = __builtin_shufflevector(lo, hi,
      0,1,2,3,4,5,6,7,8,9,10,11,12,13,14,15);
  return __builtin_bit_cast(v16bf, w);
}

__global__ __launch_bounds__(256, 1)
void scbl_wmma_bf16(const float* __restrict__ x,
                    const float* __restrict__ signs,
                    const float* __restrict__ scale_pos,
                    const float* __restrict__ scale_neg,
                    float* __restrict__ out)
{
  __shared__ __align__(16) unsigned short As[2][BM][LDP];
  __shared__ __align__(16) unsigned short Bs[2][BN][LDP];

  const int tid  = threadIdx.x;
  const int lane = tid & 31;
  const int lrow = lane & 15;
  const int lhi  = lane >> 4;          // 0 or 1 (lane half)
  const int wave = tid >> 5;           // 0..7
  const int wm   = (wave & 1) * 64;    // wave m-offset in tile
  const int wn   = (wave >> 1) * 32;   // wave n-offset in tile

  const int n0 = blockIdx.x * BN;
  const int m0 = blockIdx.y * BM;

  v8f acc[4][2];
#pragma unroll
  for (int i = 0; i < 4; ++i)
#pragma unroll
    for (int j = 0; j < 2; ++j)
      acc[i][j] = (v8f){0.f,0.f,0.f,0.f,0.f,0.f,0.f,0.f};

  // Per-thread prefetch registers: 8 float4 chunks each for A and B signs,
  // plus the (sp, sn) scale pair per chunk row.
  float4 pa[8];
  uint4  pb[8];
  float  pps[8], pns[8];

  // chunk c = tid + i*256 ; row = c>>4 ; kc = (c&15)*4  (16 float4 chunks/row)
  auto load_stage = [&](int kt) {
    const int k0 = kt * BK;
#pragma unroll
    for (int i = 0; i < 8; ++i) {
      int c   = tid + i * 256;
      int row = c >> 4;
      int kc  = (c & 15) << 2;
      pa[i] = *(const float4*)(x     + (size_t)(m0 + row) * K_TOT + k0 + kc);
      pb[i] = *(const uint4*)(signs + (size_t)(n0 + row) * K_TOT + k0 + kc);
      int g = (n0 + row) * (K_TOT / GSZ) + (k0 >> 7);  // BK=64 stays in one group
      pps[i] = scale_pos[g];
      pns[i] = scale_neg[g];
    }
  };

  auto store_stage = [&](int buf) {
#pragma unroll
    for (int i = 0; i < 8; ++i) {
      int c   = tid + i * 256;
      int row = c >> 4;
      int kc  = (c & 15) << 2;
      const float*    fa = (const float*)&pa[i];
      const unsigned* sb = (const unsigned*)&pb[i];
      unsigned short wp  = cvt_bf16(pps[i]);
      unsigned short wn_ = cvt_bf16(-pns[i]);
      v4u16 av, bv;
#pragma unroll
      for (int j = 0; j < 4; ++j) {
        av[j] = cvt_bf16(fa[j]);
        bv[j] = (sb[j] >> 31) ? wn_ : wp;   // w = sign>0 ? +sp : -sn
      }
      *(v4u16*)&As[buf][row][kc] = av;
      *(v4u16*)&Bs[buf][row][kc] = bv;
    }
  };

  auto compute_stage = [&](int buf) {
#pragma unroll
    for (int kk = 0; kk < BK; kk += 32) {
      v16bf af[4], bfv[2];
#pragma unroll
      for (int i = 0; i < 4; ++i)   // A 16x32: lane halves K {0-7,16-23}/{8-15,24-31}
        af[i] = frag_ld(&As[buf][wm + i * 16 + lrow][kk + lhi * 8], 16);
#pragma unroll
      for (int j = 0; j < 2; ++j)   // B 32x16: lane halves K {0-7,16-23}/{8-15,24-31}, col=lane&15
        bfv[j] = frag_ld(&Bs[buf][wn + j * 16 + lrow][kk + lhi * 8], 16);
#pragma unroll
      for (int i = 0; i < 4; ++i)
#pragma unroll
        for (int j = 0; j < 2; ++j)
          acc[i][j] = __builtin_amdgcn_wmma_f32_16x16x32_bf16(
              false, af[i], false, bfv[j], (short)0, acc[i][j], false, false);
    }
  };

  load_stage(0);
  store_stage(0);
  __syncthreads();

  for (int kt = 0; kt < NSTEPS; ++kt) {
    if (kt + 1 < NSTEPS) load_stage(kt + 1);   // issue global loads early
    compute_stage(kt & 1);                     // overlap WMMA with loads in flight
    if (kt + 1 < NSTEPS) {
      store_stage((kt + 1) & 1);
      __syncthreads();
    }
  }

  // Epilogue: C layout — VGPR e: row = base + e (+8 for upper lane half), col = lane&15
#pragma unroll
  for (int i = 0; i < 4; ++i) {
    int rbase = m0 + wm + i * 16 + lhi * 8;
#pragma unroll
    for (int j = 0; j < 2; ++j) {
      int col = n0 + wn + j * 16 + lrow;
#pragma unroll
      for (int e = 0; e < 8; ++e)
        out[(size_t)(rbase + e) * N_TOT + col] = acc[i][j][e];
    }
  }
}

extern "C" void kernel_launch(void* const* d_in, const int* in_sizes, int n_in,
                              void* d_out, int out_size, void* d_ws, size_t ws_size,
                              hipStream_t stream) {
  (void)in_sizes; (void)n_in; (void)out_size; (void)d_ws; (void)ws_size;
  const float* x         = (const float*)d_in[0];
  const float* signs     = (const float*)d_in[1];
  const float* scale_pos = (const float*)d_in[2];
  const float* scale_neg = (const float*)d_in[3];
  // d_in[4] = gs scalar (always 128) — baked into the kernel
  float* out = (float*)d_out;

  dim3 grid(N_TOT / BN, M_TOT / BM);   // 86 x 64
  dim3 block(256);
  scbl_wmma_bf16<<<grid, block, 0, stream>>>(x, signs, scale_pos, scale_neg, out);
}
